// Seq2Seq_32882269618855
// MI455X (gfx1250) — matrix operand order repaired and checked by reading, not compiled, not run on hardware
//
#include <hip/hip_runtime.h>
#include <math.h>

// ---------------------------------------------------------------------------
// Seq2Seq decoder for MI455X (gfx1250), wave32 + WMMA bf16.
// Strategy: run the sequential GRU on small WMMA GEMMs (weights L2-resident),
// then batch all vocab-sized projections over the full 2048 = T*B rows so the
// 400+ MB of fp32 weights stream from HBM exactly once. log-softmax over the
// 50005-vocab is a two-pass (partials -> lse -> recompute+write) scheme.
// All matrix math lowers to v_wmma_f32_16x16x32_bf16; fp32->bf16 uses native
// hardware converts; streaming weight operand is prefetched (global_prefetch).
// ---------------------------------------------------------------------------

typedef __attribute__((ext_vector_type(16))) __bf16           v16bf;
typedef __attribute__((ext_vector_type(8)))  float            v8f;
typedef __attribute__((ext_vector_type(4)))  float            v4f;

#define VOCAB   50005
#define NPAD    50048      // 391 * 128
#define CHUNKS  391
#define KSLD    8832       // 69 * 128
#define TT      64
#define BB      32
#define ROWS    2048       // TT * BB
#define FLT_BIG 3.402823466e38f
#define NEGC    3.720075976020836e-44f   // exp(-100)

__device__ __forceinline__ float sigmoidf_(float x) { return 1.f / (1.f + __expf(-x)); }

// Pack four float4s into one 16-element bf16 fragment via native converts.
__device__ __forceinline__ v16bf cvt16(v4f a, v4f b, v4f c, v4f d) {
    v16bf r;
#pragma unroll
    for (int i = 0; i < 4; ++i) {
        r[i]      = (__bf16)a[i];
        r[i + 4]  = (__bf16)b[i];
        r[i + 8]  = (__bf16)c[i];
        r[i + 12] = (__bf16)d[i];
    }
    return r;
}

// Accumulate a 16x16 fp32 tile of  C += A(16xK) * B(16xK)^T  using bf16 WMMA.
// A: row-major, points at row 0 of the 16-row M tile, leading dim lda.
// Bt: row-major [N][K] weight matrix (row n = output column n), leading dim ldb.
// n0: first output column of this tile; weight rows clamped to nmax (masked later).
__device__ __forceinline__ void wmma_acc(v8f& acc,
                                         const float* __restrict__ A, int lda, int K,
                                         const float* __restrict__ Bt, int n0, int ldb, int nmax)
{
    const int lane = threadIdx.x & 31;
    const int half = lane >> 4;
    const int lr   = lane & 15;
    const float* arow = A + (size_t)lr * (size_t)lda;
    int brow = n0 + lr; if (brow > nmax) brow = nmax;
    const float* bptr = Bt + (size_t)brow * (size_t)ldb;

    for (int k0 = 0; k0 < K; k0 += 32) {
        // ---- A fragment (16x32 bf16): lanes<16 -> K {k0..k0+7, k0+16..k0+23},
        //                               lanes>=16 -> K {k0+8..15, k0+24..31}
        const int ka = k0 + half * 8;
        v4f a0 = *(const v4f*)(arow + ka);
        v4f a1 = *(const v4f*)(arow + ka + 4);
        v4f a2 = *(const v4f*)(arow + ka + 16);
        v4f a3 = *(const v4f*)(arow + ka + 20);
        // ---- B fragment (32x16 bf16): column = lane%16; lanes<16 K {k0..k0+7, k0+16..k0+23},
        //                               lanes>=16 K {k0+8..15, k0+24..31}
        const int kb = k0 + half * 8;
        v4f b0 = *(const v4f*)(bptr + kb);
        v4f b1 = *(const v4f*)(bptr + kb + 4);
        v4f b2 = *(const v4f*)(bptr + kb + 16);
        v4f b3 = *(const v4f*)(bptr + kb + 20);

        // Prefetch the next K-block of the streaming weight operand
        // (lowers to global_prefetch_b8; speculative, dropped if OOB).
        __builtin_prefetch(bptr + kb + 32, 0, 0);

        v16bf af = cvt16(a0, a1, a2, a3);
        v16bf bf = cvt16(b0, b1, b2, b3);
        acc = __builtin_amdgcn_wmma_f32_16x16x32_bf16(false, af, false, bf,
                                                      (short)0, acc, false, false);
    }
}

// ---------------------------------------------------------------------------
// Embedding gather: emb_all[(t*32+b)][0..511] = embed_w[target[b][t]]
__global__ void k_embed_gather(const int* __restrict__ target,
                               const float* __restrict__ embed_w,
                               float* __restrict__ emb_all)
{
    int rid = blockIdx.x;            // t*32 + b
    int t = rid >> 5, b = rid & 31;
    int tok = target[b * TT + t];
    const float4* src = (const float4*)(embed_w + (size_t)tok * 512);
    float4* dst = (float4*)(emb_all + (size_t)rid * 512);
    dst[threadIdx.x] = src[threadIdx.x];   // 128 threads * float4 = 512 floats
}

// Generic small-M WMMA GEMM: C = act(A @ Bt^T + bias). Block = 8 waves, tile 16x128.
__global__ void k_gemm32(const float* __restrict__ A, int lda, int K,
                         const float* __restrict__ Bt, int ldb,
                         const float* __restrict__ bias,
                         float* __restrict__ C, int ldc, int Nvalid, int act)
{
    int w = threadIdx.x >> 5, lane = threadIdx.x & 31, half = lane >> 4, lr = lane & 15;
    int n0 = blockIdx.x * 128 + w * 16;
    int mt = blockIdx.y;
    v8f acc = {};
    wmma_acc(acc, A + (size_t)mt * 16 * lda, lda, K, Bt, n0, ldb, Nvalid - 1);
    int n = n0 + lr;
    if (n < Nvalid) {
        float bv = bias ? bias[n] : 0.f;
#pragma unroll
        for (int r = 0; r < 8; ++r) {
            float v = acc[r] + bv;
            if (act == 1) v = tanhf(v);
            C[(size_t)(mt * 16 + r + 8 * half) * ldc + n] = v;
        }
    }
}

// Dual-input GEMM: C = A1@B1t^T + A2@B2t^T (used for gate_const).
__global__ void k_gemm32_dual(const float* __restrict__ A1, int lda1, int K1,
                              const float* __restrict__ B1t, int ldb1,
                              const float* __restrict__ A2, int lda2, int K2,
                              const float* __restrict__ B2t, int ldb2,
                              float* __restrict__ C, int ldc, int Nvalid)
{
    int w = threadIdx.x >> 5, lane = threadIdx.x & 31, half = lane >> 4, lr = lane & 15;
    int n0 = blockIdx.x * 128 + w * 16;
    int mt = blockIdx.y;
    v8f acc = {};
    wmma_acc(acc, A1 + (size_t)mt * 16 * lda1, lda1, K1, B1t, n0, ldb1, Nvalid - 1);
    wmma_acc(acc, A2 + (size_t)mt * 16 * lda2, lda2, K2, B2t, n0, ldb2, Nvalid - 1);
    int n = n0 + lr;
    if (n < Nvalid) {
#pragma unroll
        for (int r = 0; r < 8; ++r)
            C[(size_t)(mt * 16 + r + 8 * half) * ldc + n] = acc[r];
    }
}

// GRU elementwise combine (torch variant): h' = (1-z)*n + z*h
__global__ void k_gru_combine(const float* __restrict__ gi, const float* __restrict__ gh,
                              const float* __restrict__ hprev, float* __restrict__ hnew)
{
    int idx = blockIdx.x * blockDim.x + threadIdx.x;
    if (idx >= BB * 1024) return;
    int b = idx >> 10, j = idx & 1023;
    const float* gib = gi + (size_t)b * 3072;
    const float* ghb = gh + (size_t)b * 3072;
    float r = sigmoidf_(gib[j] + ghb[j]);
    float z = sigmoidf_(gib[1024 + j] + ghb[1024 + j]);
    float n = tanhf(gib[2048 + j] + r * ghb[2048 + j]);
    hnew[idx] = (1.f - z) * n + z * hprev[idx];
}

// total = h_all@dec_inf_w^T + emb_all@emb_inf_w^T + emb_inf_b + ses_inf[b]; maxout pairs.
__global__ void k_total_mx(const float* __restrict__ h_all, const float* __restrict__ dec_inf_w,
                           const float* __restrict__ emb_all, const float* __restrict__ emb_inf_w,
                           const float* __restrict__ emb_inf_b, const float* __restrict__ ses_inf,
                           float* __restrict__ mx_all)
{
    __shared__ float tile[16][128];
    int w = threadIdx.x >> 5, lane = threadIdx.x & 31, half = lane >> 4, lr = lane & 15;
    int mt = blockIdx.y;                      // 0..127 row tiles
    int n0 = blockIdx.x * 128 + w * 16;       // 0..1023
    v8f acc = {};
    wmma_acc(acc, h_all  + (size_t)mt * 16 * 1024, 1024, 1024, dec_inf_w, n0, 1024, 1023);
    wmma_acc(acc, emb_all + (size_t)mt * 16 * 512,  512,  512,  emb_inf_w, n0,  512, 1023);
    int n = n0 + lr;
#pragma unroll
    for (int r = 0; r < 8; ++r) {
        int m = r + 8 * half;
        int b = (mt * 16 + m) & 31;
        tile[m][w * 16 + lr] = acc[r] + emb_inf_b[n] + ses_inf[(size_t)b * 1024 + n];
    }
    __syncthreads();
    for (int idx = threadIdx.x; idx < 16 * 64; idx += 256) {
        int m = idx >> 6, c = idx & 63;
        mx_all[(size_t)(mt * 16 + m) * 512 + blockIdx.x * 64 + c] =
            fmaxf(tile[m][2 * c], tile[m][2 * c + 1]);
    }
}

// Pass A: per-(row, chunk) partial (max, sum-exp) of logits = mx_all @ embed_w^T
__global__ void k_logits_reduce(const float* __restrict__ mx_all,
                                const float* __restrict__ embed_w,
                                float* __restrict__ bMax, float* __restrict__ bSum)
{
    __shared__ float sm[16][8], ss[16][8];
    int w = threadIdx.x >> 5, lane = threadIdx.x & 31, half = lane >> 4, lr = lane & 15;
    int mt = blockIdx.y, chunk = blockIdx.x;
    int n0 = chunk * 128 + w * 16;
    v8f acc = {};
    wmma_acc(acc, mx_all + (size_t)mt * 16 * 512, 512, 512, embed_w, n0, 512, VOCAB - 1);
    bool valid = (n0 + lr) < VOCAB;
#pragma unroll
    for (int r = 0; r < 8; ++r) {
        float v = valid ? acc[r] : -FLT_BIG;
        float m = v;
#pragma unroll
        for (int off = 1; off < 16; off <<= 1) m = fmaxf(m, __shfl_xor(m, off, 32));
        float e = valid ? __expf(v - m) : 0.f;
#pragma unroll
        for (int off = 1; off < 16; off <<= 1) e += __shfl_xor(e, off, 32);
        if (lr == 0) { sm[r + 8 * half][w] = m; ss[r + 8 * half][w] = e; }
    }
    __syncthreads();
    if (threadIdx.x < 16) {
        int m = threadIdx.x;
        float M = -FLT_BIG;
        for (int i = 0; i < 8; ++i) M = fmaxf(M, sm[m][i]);
        float S = 0.f;
        for (int i = 0; i < 8; ++i) { float s = ss[m][i]; if (s > 0.f) S += s * __expf(sm[m][i] - M); }
        bMax[(size_t)(mt * 16 + m) * CHUNKS + chunk] = M;
        bSum[(size_t)(mt * 16 + m) * CHUNKS + chunk] = S;
    }
}

// lse[row] = M + log(sum_i S_i * exp(M_i - M))
__global__ void k_lse(const float* __restrict__ bMax, const float* __restrict__ bSum,
                      float* __restrict__ lse)
{
    __shared__ float red[128];
    int row = blockIdx.x, tid = threadIdx.x;
    float lm = -FLT_BIG;
    for (int i = tid; i < CHUNKS; i += 128) lm = fmaxf(lm, bMax[(size_t)row * CHUNKS + i]);
    red[tid] = lm; __syncthreads();
    for (int s = 64; s > 0; s >>= 1) { if (tid < s) red[tid] = fmaxf(red[tid], red[tid + s]); __syncthreads(); }
    float M = red[0]; __syncthreads();
    float ls = 0.f;
    for (int i = tid; i < CHUNKS; i += 128) {
        float sv = bSum[(size_t)row * CHUNKS + i];
        if (sv > 0.f) ls += sv * __expf(bMax[(size_t)row * CHUNKS + i] - M);
    }
    red[tid] = ls; __syncthreads();
    for (int s = 64; s > 0; s >>= 1) { if (tid < s) red[tid] += red[tid + s]; __syncthreads(); }
    if (tid == 0) lse[row] = M + __logf(red[0]);
}

// know_probs: log_softmax over [NEG*10, ks[0..8783], NEG*41211]
__global__ void k_know_probs(const float* __restrict__ ks, float* __restrict__ knowP)
{
    __shared__ float red[256];
    int b = blockIdx.x, tid = threadIdx.x;
    const float* row = ks + (size_t)b * KSLD;
    float lm = NEGC;
    for (int i = tid; i < 8784; i += 256) lm = fmaxf(lm, row[i]);
    red[tid] = lm; __syncthreads();
    for (int s = 128; s > 0; s >>= 1) { if (tid < s) red[tid] = fmaxf(red[tid], red[tid + s]); __syncthreads(); }
    float M = red[0]; __syncthreads();
    float ls = 0.f;
    for (int i = tid; i < 8784; i += 256) ls += __expf(row[i] - M);
    if (tid == 0) ls += (float)(VOCAB - 8784) * __expf(NEGC - M);
    red[tid] = ls; __syncthreads();
    for (int s = 128; s > 0; s >>= 1) { if (tid < s) red[tid] += red[tid + s]; __syncthreads(); }
    float lseK = M + __logf(red[0]);
    for (int v = tid; v < VOCAB; v += 256) {
        float sc = (v >= 10 && v < 10 + 8784) ? row[v - 10] : NEGC;
        knowP[(size_t)b * NPAD + v] = sc - lseK;
    }
}

// Pass B: recompute logits + l3 + l4, gate, mix with know_probs, LDS-staged coalesced store.
__global__ void k_final(const float* __restrict__ mx_all, const float* __restrict__ embed_w,
                        const float* __restrict__ h_all,  const float* __restrict__ W3,
                        const float* __restrict__ emb_all, const float* __restrict__ W4,
                        const float* __restrict__ gateC,  const float* __restrict__ knowP,
                        const float* __restrict__ lse,    float* __restrict__ out)
{
    __shared__ float tile[16][128];
    int w = threadIdx.x >> 5, lane = threadIdx.x & 31, half = lane >> 4, lr = lane & 15;
    int mt = blockIdx.y, chunk = blockIdx.x;
    int n0 = chunk * 128 + w * 16;
    v8f accL = {}, acc3 = {}, acc4 = {};
    wmma_acc(accL, mx_all + (size_t)mt * 16 * 512,  512,  512,  embed_w, n0, 512,  VOCAB - 1);
    wmma_acc(acc3, h_all  + (size_t)mt * 16 * 1024, 1024, 1024, W3,      n0, 1024, VOCAB - 1);
    wmma_acc(acc4, emb_all + (size_t)mt * 16 * 512, 512,  512,  W4,      n0, 512,  VOCAB - 1);
    int vc = n0 + lr; if (vc > VOCAB - 1) vc = VOCAB - 1;
#pragma unroll
    for (int r = 0; r < 8; ++r) {
        int m = r + 8 * half;
        int row = mt * 16 + m;
        int b = row & 31;
        float hid = accL[r] - lse[row];
        float g = sigmoidf_(gateC[(size_t)b * NPAD + vc] + acc3[r] + acc4[r]);
        tile[m][w * 16 + lr] = g * hid + (1.f - g) * knowP[(size_t)b * NPAD + vc];
    }
    __syncthreads();
#pragma unroll
    for (int it = 0; it < 8; ++it) {
        int m = it * 2 + (threadIdx.x >> 7);
        int col = threadIdx.x & 127;
        int v = chunk * 128 + col;
        if (v < VOCAB) {
            int row = mt * 16 + m;
            int b = row & 31, t = row >> 5;
            out[((size_t)b * TT + t) * VOCAB + v] = tile[m][col];
        }
    }
}

// ---------------------------------------------------------------------------
extern "C" void kernel_launch(void* const* d_in, const int* in_sizes, int n_in,
                              void* d_out, int out_size, void* d_ws, size_t ws_size,
                              hipStream_t stream)
{
    (void)in_sizes; (void)n_in; (void)out_size; (void)ws_size;
    const float* ses       = (const float*)d_in[0];   // (32,1,1024)
    const float* know      = (const float*)d_in[1];   // (32,1,1024)
    const int*   target    = (const int*)  d_in[2];   // (32,64)
    const float* embed_w   = (const float*)d_in[3];   // (50005,512)
    const float* W_ih      = (const float*)d_in[4];   // (3072,512)
    const float* W_hh      = (const float*)d_in[5];   // (3072,1024)
    const float* b_ih      = (const float*)d_in[6];
    const float* b_hh      = (const float*)d_in[7];
    const float* ses2dec_w = (const float*)d_in[8];   // (1024,1024)
    const float* ses2dec_b = (const float*)d_in[9];
    const float* dec_inf_w = (const float*)d_in[10];  // (1024,1024)
    const float* ses_inf_w = (const float*)d_in[11];  // (1024,1024)
    const float* emb_inf_w = (const float*)d_in[12];  // (1024,512)
    const float* emb_inf_b = (const float*)d_in[13];
    const float* W1        = (const float*)d_in[14];  // (50005,1024)
    const float* W2        = (const float*)d_in[15];  // (50005,1024)
    const float* W3        = (const float*)d_in[16];  // (50005,1024)
    const float* W4        = (const float*)d_in[17];  // (50005,512)
    const float* know_vote = (const float*)d_in[18];  // (8785,1024)
    float* out = (float*)d_out;

    // Workspace carve-up (~38 MB of fp32)
    float* ws      = (float*)d_ws;
    float* emb_all = ws;  ws += (size_t)ROWS * 512;     // (t*32+b, 512)
    float* h_all   = ws;  ws += (size_t)ROWS * 1024;    // (t*32+b, 1024)
    float* h0      = ws;  ws += (size_t)BB * 1024;
    float* ses_inf = ws;  ws += (size_t)BB * 1024;
    float* mx_all  = ws;  ws += (size_t)ROWS * 512;
    float* gi      = ws;  ws += (size_t)BB * 3072;
    float* gh      = ws;  ws += (size_t)BB * 3072;
    float* gateC   = ws;  ws += (size_t)BB * NPAD;
    float* ksbuf   = ws;  ws += (size_t)BB * KSLD;
    float* knowP   = ws;  ws += (size_t)BB * NPAD;
    float* bMax    = ws;  ws += (size_t)ROWS * CHUNKS;
    float* bSum    = ws;  ws += (size_t)ROWS * CHUNKS;
    float* lse     = ws;  ws += (size_t)ROWS;

    // --- precompute ---
    k_embed_gather<<<ROWS, 128, 0, stream>>>(target, embed_w, emb_all);
    k_gemm32<<<dim3(8, 2), 256, 0, stream>>>(ses, 1024, 1024, ses2dec_w, 1024,
                                             ses2dec_b, h0, 1024, 1024, 1);        // tanh
    k_gemm32<<<dim3(8, 2), 256, 0, stream>>>(ses, 1024, 1024, ses_inf_w, 1024,
                                             nullptr, ses_inf, 1024, 1024, 0);
    k_gemm32_dual<<<dim3(CHUNKS, 2), 256, 0, stream>>>(ses, 1024, 1024, W1, 1024,
                                                       know, 1024, 1024, W2, 1024,
                                                       gateC, NPAD, VOCAB);
    k_gemm32<<<dim3(69, 2), 256, 0, stream>>>(know, 1024, 1024, know_vote, 1024,
                                              nullptr, ksbuf, KSLD, 8785, 0);
    k_know_probs<<<BB, 256, 0, stream>>>(ksbuf, knowP);

    // --- sequential GRU (weights 18 MB -> L2 resident across 64 steps) ---
    const float* hprev = h0;
    for (int t = 0; t < TT; ++t) {
        k_gemm32<<<dim3(24, 2), 256, 0, stream>>>(emb_all + (size_t)t * BB * 512, 512, 512,
                                                  W_ih, 512, b_ih, gi, 3072, 3072, 0);
        k_gemm32<<<dim3(24, 2), 256, 0, stream>>>(hprev, 1024, 1024,
                                                  W_hh, 1024, b_hh, gh, 3072, 3072, 0);
        float* hnew = h_all + (size_t)t * BB * 1024;
        k_gru_combine<<<128, 256, 0, stream>>>(gi, gh, hprev, hnew);
        hprev = hnew;
    }

    // --- batched vocab-sized passes (each big weight streamed once) ---
    k_total_mx<<<dim3(8, 128), 256, 0, stream>>>(h_all, dec_inf_w, emb_all, emb_inf_w,
                                                 emb_inf_b, ses_inf, mx_all);
    k_logits_reduce<<<dim3(CHUNKS, 128), 256, 0, stream>>>(mx_all, embed_w, bMax, bSum);
    k_lse<<<ROWS, 128, 0, stream>>>(bMax, bSum, lse);
    k_final<<<dim3(CHUNKS, 128), 256, 0, stream>>>(mx_all, embed_w, h_all, W3,
                                                   emb_all, W4, gateC, knowP, lse, out);
}
